// PredictiveCodingNet_42417097015531
// MI455X (gfx1250) — matrix operand order repaired and checked by reading, not compiled, not run on hardware
//
#include <hip/hip_runtime.h>
#include <stdint.h>

typedef __attribute__((ext_vector_type(16))) _Float16 v16h;
typedef __attribute__((ext_vector_type(8)))  float    v8f;
typedef __attribute__((ext_vector_type(4)))  float    v4f;

#define IN_DIM 784
#define H1 128
#define H2 64
#define H3 32
#define OUT_DIM 10
#define NSTEPS 20
#define KT 25        // K tiles of 32 (784 padded to 800)
#define NT 15        // N tiles of 16 (234 padded to 240)
#define NPAD 240
#define CHUNK_BYTES 15360u   // NT * 512 halves * 2B per K-chunk
#define STAGE_BYTES 16384u   // staged per chunk (1024 lanes x 16B), overfetch ok
#define BUF_HALVES 8192      // 16384 B per LDS buffer

// workspace layout (bytes):
//   [0, 384000)        : B matrix, f16, WMMA-fragment-swizzled  (KT*NT*512 halves)
//   [384000, 384960)   : q vector, f32, 240 entries
//   [385024, 507904)   : G matrix, f32, 128 x 240
#define WS_B_BYTES (KT * NT * 512 * 2)
#define WS_Q_OFF   384000
#define WS_G_OFF   385024

// ---------------------------------------------------------------------------
// Stage A1: compute G = 0.2 * sum_{k=0}^{19} T^k restricted to rows 0..127,
// and q = final bias/init state after 20 steps. T is the affine step matrix:
//   p1' = 0.8 p1 + 0.2 (c1 + p1 L1)
//   p2' = 0.8 p2 + 0.2 (p1 W2^T + c2 + p2 L2)
//   p3' = 0.8 p3 + 0.2 (p2 W3^T + c3 + p3 L3)
//   p4' = 0.8 p4 + 0.2 (p3 W4^T + c4)
// For G rows: c=0, init m1 = e_r.  For q: c=b_i, init q4 = 0.1.
// ---------------------------------------------------------------------------
__global__ __launch_bounds__(128) void pc_precompute(
    const float* __restrict__ W2, const float* __restrict__ W3,
    const float* __restrict__ W4,
    const float* __restrict__ L1, const float* __restrict__ L2,
    const float* __restrict__ L3,
    const float* __restrict__ b1, const float* __restrict__ b2,
    const float* __restrict__ b3, const float* __restrict__ b4,
    float* __restrict__ G, float* __restrict__ q) {
  __shared__ float m1[H1];
  __shared__ float m2[H2];
  __shared__ float m3[H3];
  __shared__ float m4[16];

  const int tid = threadIdx.x;
  const int r = blockIdx.x;
  const bool qmode = (r == H1);
  const int jj = tid - 64;  // role index for threads >= 64
  const int j4 = tid - 96;

  // init state
  if (tid < 64) {
    m1[tid]      = (!qmode && tid == r) ? 1.f : 0.f;
    m1[tid + 64] = (!qmode && (tid + 64) == r) ? 1.f : 0.f;
  } else {
    m2[jj] = 0.f;
    if (tid < 96) m3[jj] = 0.f;
    else if (tid < 112) m4[j4] = (qmode && j4 < OUT_DIM) ? 0.1f : 0.f;
  }
  __syncthreads();

  float g1a = 0.f, g1b = 0.f, g2 = 0.f, g3 = 0.f, g4 = 0.f;

  for (int step = 0; step < NSTEPS; ++step) {
    float n1a = 0.f, n1b = 0.f, n2 = 0.f, n3 = 0.f, n4 = 0.f;
    if (tid < 64) {
      g1a += m1[tid];
      g1b += m1[tid + 64];
      float a0 = qmode ? b1[tid] : 0.f;
      float a1 = qmode ? b1[tid + 64] : 0.f;
#pragma unroll 8
      for (int k = 0; k < H1; ++k) {
        const float s = m1[k];
        a0 += s * L1[k * H1 + tid];
        a1 += s * L1[k * H1 + tid + 64];
      }
      n1a = 0.8f * m1[tid] + 0.2f * a0;
      n1b = 0.8f * m1[tid + 64] + 0.2f * a1;
    } else {
      g2 += m2[jj];
      float a2 = qmode ? b2[jj] : 0.f;
#pragma unroll 8
      for (int k = 0; k < H1; ++k) a2 += m1[k] * W2[jj * H1 + k];
#pragma unroll 8
      for (int k = 0; k < H2; ++k) a2 += m2[k] * L2[k * H2 + jj];
      n2 = 0.8f * m2[jj] + 0.2f * a2;
      if (tid < 96) {
        g3 += m3[jj];
        float a3 = qmode ? b3[jj] : 0.f;
#pragma unroll 8
        for (int k = 0; k < H2; ++k) a3 += m2[k] * W3[jj * H2 + k];
#pragma unroll 8
        for (int k = 0; k < H3; ++k) a3 += m3[k] * L3[k * H3 + jj];
        n3 = 0.8f * m3[jj] + 0.2f * a3;
      } else if (j4 < OUT_DIM) {
        g4 += m4[j4];
        float a4 = qmode ? b4[j4] : 0.f;
#pragma unroll
        for (int k = 0; k < H3; ++k) a4 += m3[k] * W4[j4 * H3 + k];
        n4 = 0.8f * m4[j4] + 0.2f * a4;
      }
    }
    __syncthreads();
    if (tid < 64) {
      m1[tid] = n1a;
      m1[tid + 64] = n1b;
    } else {
      m2[jj] = n2;
      if (tid < 96) m3[jj] = n3;
      else if (j4 < OUT_DIM) m4[j4] = n4;
    }
    __syncthreads();
  }

  if (qmode) {
    // q = final state after 20 steps
    if (tid < 64) {
      q[tid] = m1[tid];
      q[tid + 64] = m1[tid + 64];
    } else {
      q[H1 + jj] = m2[jj];
      if (tid < 96) q[192 + jj] = m3[jj];
      else if (tid < 112) q[224 + j4] = (j4 < OUT_DIM) ? m4[j4] : 0.f;
    }
  } else {
    // G row r = 0.2 * accumulated sum
    float* Gr = G + r * NPAD;
    if (tid < 64) {
      Gr[tid] = 0.2f * g1a;
      Gr[tid + 64] = 0.2f * g1b;
    } else {
      Gr[H1 + jj] = 0.2f * g2;
      if (tid < 96) Gr[192 + jj] = 0.2f * g3;
      else if (j4 < OUT_DIM) Gr[224 + j4] = 0.2f * g4;
    }
  }
}

// ---------------------------------------------------------------------------
// Stage A2: Pcat[r,n] = sum_k W1[k,r] * G[k,n]  (784 x 234, f32 accumulation),
// stored as f16 directly in the WMMA B-fragment swizzle:
//   frag(ktile,ntile): lane = ((k%16)/8)*16 + (n%16), half index h = k%8 + 8*((k%32)/16)
// ---------------------------------------------------------------------------
__global__ __launch_bounds__(256) void pc_expand(
    const float* __restrict__ W1, const float* __restrict__ G,
    _Float16* __restrict__ Bws) {
  const int gid = blockIdx.x * 256 + threadIdx.x;
  const int r = gid / NPAD;
  const int n = gid % NPAD;
  if (r >= IN_DIM || n >= 234) return;
  float acc = 0.f;
#pragma unroll 8
  for (int k = 0; k < H1; ++k) acc += W1[k * IN_DIM + r] * G[k * NPAD + n];
  const int ktile = r >> 5;
  const int h = (r & 7) + ((r >> 4) & 1) * 8;
  const int lane = ((r >> 3) & 1) * 16 + (n & 15);
  const int idx = ((ktile * NT + (n >> 4)) * 32 + lane) * 16 + h;
  Bws[idx] = (_Float16)acc;
}

// ---------------------------------------------------------------------------
// Stage B: out[row, n] = sum_k x[row,k] * Pcat[k,n] + q[n]
// 512 blocks x 256 threads (8 waves). Wave w owns M-rows [w*16, w*16+16),
// all 15 N-tiles. 25 K-chunks of 32. B fragments DMA'd to LDS with
// global_load_async_to_lds_b128 (ASYNCcnt), triple-buffered, one barrier
// per chunk; x loads software-pipelined one chunk ahead.
// ---------------------------------------------------------------------------
__global__ __launch_bounds__(256) void pc_gemm(
    const float* __restrict__ x, const _Float16* __restrict__ Bws,
    const float* __restrict__ q, float* __restrict__ out, const long batch) {
  __shared__ __align__(16) _Float16 ldsB[3 * BUF_HALVES];
  const int tid = threadIdx.x;
  const int wave = tid >> 5;
  const int lane = tid & 31;
  const int g = lane >> 4;   // lane group (0/1)
  const int nloc = lane & 15;
  // A fragment: lane holds row M = lane%16 of this wave's 16-row tile
  const long rowA = (long)blockIdx.x * 128 + wave * 16 + nloc;
  const float* xrow = x + rowA * IN_DIM;

  const uint32_t ldsBase = (uint32_t)(uintptr_t)(&ldsB[0]);
  const unsigned long long bws64 = (unsigned long long)Bws;

  // issue 4 async b128 loads per lane: stages STAGE_BYTES of B into buffer BUF
#define ISSUE_STAGE(KC, BUF)                                               \
  do {                                                                     \
    const uint32_t src0_ = (uint32_t)(KC)*CHUNK_BYTES;                     \
    const uint32_t dst0_ = ldsBase + (uint32_t)(BUF)*STAGE_BYTES;          \
    _Pragma("unroll") for (int j_ = 0; j_ < 4; ++j_) {                     \
      const uint32_t off_ = ((uint32_t)tid + (uint32_t)j_ * 256u) * 16u;   \
      const uint32_t vs_ = src0_ + off_;                                   \
      const uint32_t vd_ = dst0_ + off_;                                   \
      asm volatile("global_load_async_to_lds_b128 %0, %1, %2"              \
                   ::"v"(vd_), "v"(vs_), "s"(bws64)                        \
                   : "memory");                                            \
    }                                                                      \
  } while (0)

  ISSUE_STAGE(0, 0);

  v8f acc[NT];
  const v8f zero8 = {0.f, 0.f, 0.f, 0.f, 0.f, 0.f, 0.f, 0.f};
#pragma unroll
  for (int i = 0; i < NT; ++i) acc[i] = zero8;

  // x loads for chunk 0 (fully in-bounds)
  v4f xa, xb, xc, xd;
  {
    const float* p = xrow + g * 8;
    xa = __builtin_nontemporal_load((const v4f*)p);
    xb = __builtin_nontemporal_load((const v4f*)(p + 4));
    xc = __builtin_nontemporal_load((const v4f*)(p + 16));
    xd = __builtin_nontemporal_load((const v4f*)(p + 20));
  }
  bool hiok = true;

  for (int kc = 0; kc < KT; ++kc) {
    if (kc + 1 < KT) {
      ISSUE_STAGE(kc + 1, (kc + 1) % 3);
      asm volatile("s_wait_asynccnt 4" ::: "memory");  // oldest 4 (chunk kc) done
    } else {
      asm volatile("s_wait_asynccnt 0" ::: "memory");
    }
    __syncthreads();  // all waves' slices of chunk kc are in LDS

    // build A fragment (lane L, half h -> K = kc*32 + (L/16)*8 + h (+8 if h>=8))
    v16h a;
    a[0] = (_Float16)xa[0]; a[1] = (_Float16)xa[1];
    a[2] = (_Float16)xa[2]; a[3] = (_Float16)xa[3];
    a[4] = (_Float16)xb[0]; a[5] = (_Float16)xb[1];
    a[6] = (_Float16)xb[2]; a[7] = (_Float16)xb[3];
    if (hiok) {
      a[8]  = (_Float16)xc[0]; a[9]  = (_Float16)xc[1];
      a[10] = (_Float16)xc[2]; a[11] = (_Float16)xc[3];
      a[12] = (_Float16)xd[0]; a[13] = (_Float16)xd[1];
      a[14] = (_Float16)xd[2]; a[15] = (_Float16)xd[3];
    } else {  // K padding 784..799 (B side zeroed too)
      a[8] = (_Float16)0.f;  a[9] = (_Float16)0.f;
      a[10] = (_Float16)0.f; a[11] = (_Float16)0.f;
      a[12] = (_Float16)0.f; a[13] = (_Float16)0.f;
      a[14] = (_Float16)0.f; a[15] = (_Float16)0.f;
    }

    // pipeline: issue x loads for the next chunk now, convert next iteration
    if (kc + 1 < KT) {
      const int k0 = (kc + 1) * 32 + g * 8;
      const float* p = xrow + k0;
      xa = __builtin_nontemporal_load((const v4f*)p);
      xb = __builtin_nontemporal_load((const v4f*)(p + 4));
      hiok = (k0 + 23 < IN_DIM);
      if (hiok) {
        xc = __builtin_nontemporal_load((const v4f*)(p + 16));
        xd = __builtin_nontemporal_load((const v4f*)(p + 20));
      }
    }

    // WMMA chain with B-fragment register preload (overlap ds_load latency)
    const v16h* bptr = (const v16h*)(ldsB + (kc % 3) * BUF_HALVES);
    v16h bcur = bptr[lane];
#pragma unroll
    for (int nt = 0; nt < NT; ++nt) {
      v16h bnxt;
      if (nt + 1 < NT) bnxt = bptr[(nt + 1) * 32 + lane];
      acc[nt] = __builtin_amdgcn_wmma_f32_16x16x32_f16(
          false, a, false, bcur, (short)0, acc[nt], false, false);
      bcur = bnxt;
    }
  }
#undef ISSUE_STAGE

  // Epilogue. C layout: lane L, VGPR v -> M = v + 8*(L/16), N = L%16.
  float* s1 = out;
  float* s2 = out + batch * H1;
  float* s3 = s2 + batch * H2;
  float* s4 = s3 + batch * H3;
  const long rowBase = (long)blockIdx.x * 128 + wave * 16 + g * 8;
#pragma unroll
  for (int nt = 0; nt < NT; ++nt) {
    const int n = nt * 16 + nloc;
    float* base;
    int stride, cn;
    if (nt < 8)       { base = s1; stride = H1;      cn = n; }
    else if (nt < 12) { base = s2; stride = H2;      cn = n - 128; }
    else if (nt < 14) { base = s3; stride = H3;      cn = n - 192; }
    else              { base = s4; stride = OUT_DIM; cn = n - 224; }
    if (nt < 14 || cn < OUT_DIM) {
      const float qv = q[n];
#pragma unroll
      for (int v = 0; v < 8; ++v) {
        __builtin_nontemporal_store(acc[nt][v] + qv,
                                    &base[(rowBase + v) * stride + cn]);
      }
    }
  }
}

extern "C" void kernel_launch(void* const* d_in, const int* in_sizes, int n_in,
                              void* d_out, int out_size, void* d_ws,
                              size_t ws_size, hipStream_t stream) {
  const float* x  = (const float*)d_in[0];
  const float* W1 = (const float*)d_in[1];
  const float* W2 = (const float*)d_in[2];
  const float* W3 = (const float*)d_in[3];
  const float* W4 = (const float*)d_in[4];
  const float* L1 = (const float*)d_in[5];
  const float* L2 = (const float*)d_in[6];
  const float* L3 = (const float*)d_in[7];
  const float* b1 = (const float*)d_in[8];
  const float* b2 = (const float*)d_in[9];
  const float* b3 = (const float*)d_in[10];
  const float* b4 = (const float*)d_in[11];
  const long batch = (long)(in_sizes[0] / IN_DIM);

  char* ws = (char*)d_ws;
  _Float16* Bws = (_Float16*)ws;
  float* q = (float*)(ws + WS_Q_OFF);
  float* G = (float*)(ws + WS_G_OFF);

  // zero B workspace: provides K-padding rows (784..799) and N-padding cols
  hipMemsetAsync(Bws, 0, (size_t)WS_B_BYTES, stream);

  pc_precompute<<<H1 + 1, 128, 0, stream>>>(W2, W3, W4, L1, L2, L3,
                                            b1, b2, b3, b4, G, q);

  pc_expand<<<(IN_DIM * NPAD + 255) / 256, 256, 0, stream>>>(W1, G, Bws);

  pc_gemm<<<(int)(batch / 128), 256, 0, stream>>>(x, Bws, q, (float*)d_out,
                                                  batch);
}
